// Attention_59897613910268
// MI455X (gfx1250) — matrix operand order repaired and checked by reading, not compiled, not run on hardware
//
#include <hip/hip_runtime.h>

typedef __attribute__((ext_vector_type(16))) __bf16 v16bf;
typedef __attribute__((ext_vector_type(8)))  float  v8f;

#define DIM  768
#define NH   12
#define HD   64
#define SEQ  256
#define NTOK 16384            // 2*8*4*256
#define WELEM (DIM*DIM)

static __device__ __forceinline__ unsigned short bf16b(float f) {
    __bf16 h = (__bf16)f;
    return __builtin_bit_cast(unsigned short, h);
}
static __device__ __forceinline__ unsigned pack2(float a, float b) {
    return (unsigned)bf16b(a) | ((unsigned)bf16b(b) << 16);
}

union ABFrag { uint4 u[2]; v16bf v; };

// A fragment (16-bit): lanes 0-15 hold K = koff+0..7 and koff+16..23;
// lanes 16-31 hold K = koff+8..15 and koff+24..31  (per CDNA5 ISA 7.12.2)
static __device__ __forceinline__ v16bf load_afrag(const unsigned short* row, int koff, int lhi) {
    ABFrag r;
    r.u[0] = *(const uint4*)(row + koff +      8 * lhi);
    r.u[1] = *(const uint4*)(row + koff + 16 + 8 * lhi);
    return r.v;
}
// B fragment (16-bit): lane holds 16 contiguous K values for its column
static __device__ __forceinline__ v16bf load_bfrag(const unsigned short* p) {
    ABFrag r;
    r.u[0] = ((const uint4*)p)[0];
    r.u[1] = ((const uint4*)p)[1];
    return r.v;
}
static __device__ __forceinline__ v8f wmma_bf16(v16bf a, v16bf b, v8f c) {
    return __builtin_amdgcn_wmma_f32_16x16x32_bf16(false, a, false, b, (short)0, c, false, false);
}
static __device__ __forceinline__ v8f zero8() {
    v8f z = {0.f, 0.f, 0.f, 0.f, 0.f, 0.f, 0.f, 0.f};
    return z;
}

// Async cache->LDS copy (CDNA5 GLOBAL_LOAD_ASYNC_TO_LDS_B128, ASYNCcnt-tracked).
// ldsaddr = low 32 bits of a generic pointer to LDS (the LDS byte address).
static __device__ __forceinline__ void async_ld_lds_b128(unsigned ldsaddr, const void* gaddr) {
    asm volatile("global_load_async_to_lds_b128 %0, %1, off"
                 :: "v"(ldsaddr), "v"((unsigned long long)(uintptr_t)gaddr)
                 : "memory");
}
static __device__ __forceinline__ void wait_asynccnt0() {
    asm volatile("s_wait_asynccnt 0x0" ::: "memory");
}

// ---------------- fp32 -> bf16 conversion ----------------
__global__ void cvt_f32_bf16(const float* __restrict__ src,
                             unsigned short* __restrict__ dst, int n4) {
    int i = blockIdx.x * blockDim.x + threadIdx.x;
    if (i < n4) {
        float4 f = ((const float4*)src)[i];
        uint2 o;
        o.x = pack2(f.x, f.y);
        o.y = pack2(f.z, f.w);
        ((uint2*)dst)[i] = o;
    }
}

// ---------------- fused QKV projection ----------------
// grid: (NTOK/128, NH, 3 {q,k,v}), block: 256 (8 waves)
// wave computes 16 rows x 64 cols, K-loop over 768 in steps of 32.
__global__ void __launch_bounds__(256)
qkv_proj(const unsigned short* __restrict__ xb,
         const unsigned short* __restrict__ wb,   // Wq,Wk,Wv contiguous, row-major [768][768]
         const float* __restrict__ bq, const float* __restrict__ bk,
         const float* __restrict__ bv,
         unsigned short* __restrict__ Qw,   // [bt][h][p][64]
         unsigned short* __restrict__ Kw,   // [bt][h][p][64]
         unsigned short* __restrict__ Vtw)  // [bt][h][d][256]
{
    const int lane = threadIdx.x & 31;
    const int wave = threadIdx.x >> 5;
    const int llo  = lane & 15, lhi = lane >> 4;
    const int m0   = blockIdx.x * 128 + wave * 16;
    const int h    = blockIdx.y;
    const int proj = blockIdx.z;

    const unsigned short* W    = wb + (size_t)proj * WELEM;
    const float*          bias = (proj == 0) ? bq : (proj == 1) ? bk : bv;

    v8f acc[4];
#pragma unroll
    for (int nt = 0; nt < 4; ++nt) acc[nt] = zero8();

    const unsigned short* xrow = xb + (size_t)(m0 + llo) * DIM;
    for (int k0 = 0; k0 < DIM; k0 += 32) {
        v16bf a = load_afrag(xrow, k0, lhi);      // x rows (M), K contiguous
#pragma unroll
        for (int nt = 0; nt < 4; ++nt) {
            int n = h * HD + nt * 16 + llo;        // output feature = W row
            v16bf b = load_afrag(W + (size_t)n * DIM, k0, lhi);
            acc[nt] = wmma_bf16(a, b, acc[nt]);
        }
    }

    const int bt    = m0 >> 8;
    const int pbase = (m0 & 255) + 8 * lhi;

    if (proj < 2) {
        unsigned short* dst = (proj == 0) ? Qw : Kw;
#pragma unroll
        for (int nt = 0; nt < 4; ++nt) {
            int d = nt * 16 + llo;
            float bn = bias[h * HD + d];
#pragma unroll
            for (int r = 0; r < 8; ++r) {
                int p = pbase + r;
                dst[(((size_t)(bt * NH + h) * SEQ + p) * HD) + d] =
                    bf16b(acc[nt][r] + bn);
            }
        }
    } else {
        // V stored transposed: [bt][h][d][p]; 8 consecutive p per lane -> b128 store
#pragma unroll
        for (int nt = 0; nt < 4; ++nt) {
            int d = nt * 16 + llo;
            float bn = bias[h * HD + d];
            uint4 o;
            o.x = pack2(acc[nt][0] + bn, acc[nt][1] + bn);
            o.y = pack2(acc[nt][2] + bn, acc[nt][3] + bn);
            o.z = pack2(acc[nt][4] + bn, acc[nt][5] + bn);
            o.w = pack2(acc[nt][6] + bn, acc[nt][7] + bn);
            *(uint4*)(Vtw + ((size_t)(bt * NH + h) * HD + d) * SEQ + pbase) = o;
        }
    }
}

// ---------------- attention ----------------
// grid: 64*12 blocks (one per batch-tile x head), block: 512 (16 waves).
// Each wave owns 16 queries. Computes S^T = K*Q^T so softmax is in-lane and
// P^T lands directly in the A-fragment layout of the PV WMMA.
// K and V^T tiles (64 KB) are staged cache->LDS with async-to-LDS DMA.
__global__ void __launch_bounds__(512)
attn(const unsigned short* __restrict__ Qw,
     const unsigned short* __restrict__ Kw,
     const unsigned short* __restrict__ Vtw,
     float* __restrict__ out)
{
    __shared__ unsigned short Klds[SEQ * HD];   // 32 KB
    __shared__ unsigned short Vlds[HD * SEQ];   // 32 KB (V transposed: [d][p])

    const int lane = threadIdx.x & 31;
    const int wave = threadIdx.x >> 5;
    const int llo  = lane & 15, lhi = lane >> 4;
    const int bh   = blockIdx.x;
    const int bt   = bh / NH, h = bh % NH;

    const int q0 = wave * 16;
    const unsigned short* qrow = Qw + (size_t)bh * (SEQ * HD) + (q0 + llo) * HD;
    __builtin_prefetch(qrow, 0, 3);   // global_prefetch_b8: Q strip in flight

    { // stage K and V^T into LDS via async cache->LDS (no VGPR round-trip)
        const uint4* kg = (const uint4*)(Kw  + (size_t)bh * (SEQ * HD));
        const uint4* vg = (const uint4*)(Vtw + (size_t)bh * (HD * SEQ));
        const unsigned kbase = (unsigned)(uintptr_t)Klds;
        const unsigned vbase = (unsigned)(uintptr_t)Vlds;
#pragma unroll
        for (int i = threadIdx.x; i < (SEQ * HD) / 8; i += 512) {
            async_ld_lds_b128(kbase + i * 16, kg + i);
            async_ld_lds_b128(vbase + i * 16, vg + i);
        }
        wait_asynccnt0();      // this wave's chunks are in LDS
    }
    __syncthreads();           // all waves' chunks are in LDS

    // B fragments of Q^T (column = query, K = head dim), reused for all key tiles
    v16bf bq0 = load_afrag(qrow,  0, lhi);
    v16bf bq1 = load_afrag(qrow, 32, lhi);

    // S^T accumulators: tile t covers keys 16t..16t+15; lane holds query llo,
    // keys 16t + 8*lhi + r
    v8f s[16];
#pragma unroll
    for (int t = 0; t < 16; ++t) {
        const unsigned short* krow = Klds + (t * 16 + llo) * HD;
        v16bf a0 = load_afrag(krow, 0, lhi);
        v16bf a1 = load_afrag(krow, 32, lhi);
        v8f c = zero8();
        c = wmma_bf16(a0, bq0, c);
        c = wmma_bf16(a1, bq1, c);
        s[t] = c;
    }

    // softmax over keys (in-lane over 128 values + partner-lane exchange)
    const float scale = 0.125f;   // (768/12)^-0.5
    float mx = -3.4e38f;
#pragma unroll
    for (int t = 0; t < 16; ++t)
#pragma unroll
        for (int r = 0; r < 8; ++r) mx = fmaxf(mx, s[t][r]);
    mx = fmaxf(mx, __shfl_xor(mx, 16, 32));

    float sum = 0.f;
#pragma unroll
    for (int t = 0; t < 16; ++t)
#pragma unroll
        for (int r = 0; r < 8; ++r) {
            float e = __expf((s[t][r] - mx) * scale);
            s[t][r] = e;
            sum += e;
        }
    sum += __shfl_xor(sum, 16, 32);
    float rinv = 1.0f / sum;

    // O = P * V : P^T accumulators repack (in registers) into A fragments.
    v8f o[4];
#pragma unroll
    for (int nt = 0; nt < 4; ++nt) o[nt] = zero8();

#pragma unroll
    for (int b = 0; b < 8; ++b) {
        union { unsigned u[8]; v16bf v; } pa;
#pragma unroll
        for (int j = 0; j < 4; ++j) {
            pa.u[j]     = pack2(s[2*b][2*j]   * rinv, s[2*b][2*j+1]   * rinv);
            pa.u[4 + j] = pack2(s[2*b+1][2*j] * rinv, s[2*b+1][2*j+1] * rinv);
        }
#pragma unroll
        for (int nt = 0; nt < 4; ++nt) {
            v16bf vb = load_afrag(Vlds + (nt * 16 + llo) * SEQ, 32 * b, lhi);
            o[nt] = wmma_bf16(pa.v, vb, o[nt]);
        }
    }

    // store: row = query q0 + r + 8*lhi, col = h*64 + nt*16 + llo (coalesced f32)
    float* orow = out + (size_t)(bt * SEQ) * DIM + h * HD;
#pragma unroll
    for (int nt = 0; nt < 4; ++nt) {
        int d = nt * 16 + llo;
#pragma unroll
        for (int r = 0; r < 8; ++r) {
            int q = q0 + r + 8 * lhi;
            orow[(size_t)q * DIM + d] = o[nt][r];
        }
    }
}

// ---------------- launch ----------------
extern "C" void kernel_launch(void* const* d_in, const int* in_sizes, int n_in,
                              void* d_out, int out_size, void* d_ws, size_t ws_size,
                              hipStream_t stream) {
    const float* x  = (const float*)d_in[0];
    const float* Wq = (const float*)d_in[1];
    const float* bq = (const float*)d_in[2];
    const float* Wk = (const float*)d_in[3];
    const float* bk = (const float*)d_in[4];
    const float* Wv = (const float*)d_in[5];
    const float* bv = (const float*)d_in[6];
    float* out = (float*)d_out;

    unsigned short* ws  = (unsigned short*)d_ws;
    unsigned short* xb  = ws;                                  // 16384*768
    unsigned short* wb  = xb + (size_t)NTOK * DIM;             // 3*768*768
    unsigned short* Qw  = wb + 3 * (size_t)WELEM;              // 16384*768
    unsigned short* Kw  = Qw + (size_t)NTOK * DIM;             // 16384*768
    unsigned short* Vtw = Kw + (size_t)NTOK * DIM;             // 16384*768

    const int nx4 = NTOK * DIM / 4;
    cvt_f32_bf16<<<(nx4 + 255) / 256, 256, 0, stream>>>(x, xb, nx4);
    const int nw4 = WELEM / 4;
    cvt_f32_bf16<<<(nw4 + 255) / 256, 256, 0, stream>>>(Wq, wb, nw4);
    cvt_f32_bf16<<<(nw4 + 255) / 256, 256, 0, stream>>>(Wk, wb + WELEM, nw4);
    cvt_f32_bf16<<<(nw4 + 255) / 256, 256, 0, stream>>>(Wv, wb + 2 * (size_t)WELEM, nw4);

    qkv_proj<<<dim3(NTOK / 128, NH, 3), 256, 0, stream>>>(xb, wb, bq, bk, bv, Qw, Kw, Vtw);

    attn<<<dim3(64 * NH), 512, 0, stream>>>(Qw, Kw, Vtw, out);
}
